// VariationalAutoEncoder_6665789243494
// MI455X (gfx1250) — matrix operand order repaired and checked by reading, not compiled, not run on hardware
//
#include <hip/hip_runtime.h>
#include <hip/hip_bf16.h>

// ---------------------------------------------------------------------------
// VAE forward on MI455X (gfx1250, wave32).
// Live compute = 5-layer LSTM encoder (T=4096, B=48, H=64) + tiny heads +
// a 193MB "recon" broadcast. The LSTM is latency-bound (4096-step recurrence),
// so: one workgroup per layer, software-pipelined through L2 with
// acquire/release progress flags; weights held in VGPRs as preformatted
// v_wmma_f32_16x16x32_f16 B-fragments; [x_t|h] fused into one K=128 GEMM.
// ---------------------------------------------------------------------------

typedef _Float16 v8h  __attribute__((ext_vector_type(8)));
typedef _Float16 v16h __attribute__((ext_vector_type(16)));
typedef float    v8f  __attribute__((ext_vector_type(8)));

#define T_LEN  4096
#define BATCH  48
#define HID    64
#define GATES  256     // 4*H
#define LAYERS 5

// workspace layout (bytes)
#define FRAG_OFF   0u          // 5*16*4*32*16 halfs = 327,680 B (WMMA B-fragments)
#define W0T_OFF    327680u     // 256*256 halfs = 131,072 B (Wih0 transposed, f16)
#define PROG_OFF   458752u     // 5 ints (pipeline progress flags)
#define ENC_OFF    459008u     // 48*320 f32 = 61,440 B
#define SEQ_OFF    524288u     // 4 * (4096*48*64) f16 = 96 MB (inter-layer h streams)
#define SEQ_STRIDE ((size_t)T_LEN * BATCH * HID)   // halfs per layer stream

#define FRAG_ELEMS (5*16*4*32*16)

__device__ __forceinline__ float sigm(float x) { return 1.0f / (1.0f + __expf(-x)); }

// ---------------------------------------------------------------------------
// Prep: build per-lane-contiguous WMMA B-fragments for W = [Wih | Whh]^T
// (K=128, N=256 gates), f32->f16; transpose Wih0 -> [V][4H] f16 for the
// one-hot gather; zero the pipeline progress flags.
// B k order used by this code (A matches): lane -> N = lane%16 ; half j -> K = (lane/16)*16+j
// within each 32-wide K chunk. Layer 0 has no dense Wih (one-hot input), so
// its kc=0,1 fragments are zeros and all layers share one code path.
// ---------------------------------------------------------------------------
__global__ void prep_kernel(const float* __restrict__ wih0,
                            const float* __restrict__ wih,
                            const float* __restrict__ whh,
                            _Float16* __restrict__ frag,
                            _Float16* __restrict__ w0t,
                            int* __restrict__ progress) {
  int i = blockIdx.x * blockDim.x + threadIdx.x;
  if (i < LAYERS) progress[i] = 0;
  if (i < FRAG_ELEMS) {
    int j    = i & 15;
    int lane = (i >> 4) & 31;
    int kc   = (i >> 9) & 3;
    int nt   = (i >> 11) & 15;
    int l    = i >> 15;
    int g = nt * 16 + (lane & 15);                 // gate row (N)
    int k = kc * 32 + (lane >> 4) * 16 + j;        // K index (0..127)
    float v;
    if (k < 64) v = (l == 0) ? 0.0f : wih[(((l - 1) * GATES) + g) * HID + k];
    else        v = whh[((l * GATES) + g) * HID + (k - 64)];
    frag[i] = (_Float16)v;
  }
  int i2 = i - FRAG_ELEMS;
  if (i2 >= 0 && i2 < 256 * 256) {
    int g = i2 & 255;
    int v = i2 >> 8;
    w0t[i2] = (_Float16)wih0[g * 256 + v];         // w0t[v][g] = Wih0[g][v]
  }
}

// ---------------------------------------------------------------------------
// recon[b,t,v] = (t>=1 && v==x[b,0]) ? 1 : 0   -- pure store bandwidth,
// 193 MB ~ 8.3us at 23.3 TB/s. float4 stores (b128).
// ---------------------------------------------------------------------------
__global__ void recon_kernel(const int* __restrict__ x, float* __restrict__ out) {
  size_t i = (size_t)blockIdx.x * blockDim.x + threadIdx.x;   // float4 index
  int v4 = (int)(i & 63);
  int t  = (int)((i >> 6) & 4095);
  int b  = (int)(i >> 18);
  int xb = x[(size_t)b * T_LEN];                  // x[b,0]
  float4 r = make_float4(0.f, 0.f, 0.f, 0.f);
  if (t >= 1 && (xb >> 2) == v4) (&r.x)[xb & 3] = 1.0f;
  ((float4*)out)[i] = r;
}

// ---------------------------------------------------------------------------
// Pipelined 5-layer LSTM: block = layer, 256 threads = 8 wave32.
// Per step: gates = [x_t|h] @ [Wih|Whh]^T + b  via 24 v_wmma per wave
// (3 M-tiles x 2 N-tiles x 4 K-chunks), epilogue does gate exchange through
// LDS + sigmoid/tanh (v_exp_f32 TRANS ops co-execute with the matrix pipe).
// Inter-layer h streams are f16 in workspace (L2-resident, 24 MB/layer).
// ---------------------------------------------------------------------------
__global__ void __launch_bounds__(256, 1)
lstm_pipeline_kernel(const int* __restrict__ x,
                     const float* __restrict__ bias_all,
                     const _Float16* __restrict__ frag,
                     const _Float16* __restrict__ w0t,
                     _Float16* __restrict__ seqbuf,
                     int* __restrict__ progress,
                     float* __restrict__ enc) {
  const int tid   = threadIdx.x;
  const int lane  = tid & 31;
  const int wave  = tid >> 5;
  const int hlf   = lane >> 4;          // which 16-lane half
  const int lrow  = lane & 15;
  const int layer = blockIdx.x;

  __shared__ _Float16 xh[BATCH * 128];      // [b][0:64]=x_t, [b][64:128]=h  (f16)
  __shared__ float    gbuf[BATCH * GATES];  // pre-activation gates (f32)
  __shared__ float    cbuf[BATCH * HID];    // cell state
  __shared__ int      xb[BATCH];            // byte of x[:,t] (layer 0)

  // --- B fragments resident in VGPRs for the whole sequence (zero weight
  // refetch on the 4096-step critical path) ---
  const int nt0 = wave * 2, nt1 = wave * 2 + 1;
  v16h bfrag0[4], bfrag1[4];
#pragma unroll
  for (int kc = 0; kc < 4; ++kc) {
    bfrag0[kc] = *(const v16h*)(frag + ((((size_t)layer * 16 + nt0) * 4 + kc) * 32 + lane) * 16);
    bfrag1[kc] = *(const v16h*)(frag + ((((size_t)layer * 16 + nt1) * 4 + kc) * 32 + lane) * 16);
  }
  const float bv0 = bias_all[layer * GATES + nt0 * 16 + lrow];
  const float bv1 = bias_all[layer * GATES + nt1 * 16 + lrow];

  for (int i = tid; i < BATCH * 128; i += 256) xh[i] = (_Float16)0.0f;
  for (int i = tid; i < BATCH * HID; i += 256) cbuf[i] = 0.0f;
  __syncthreads();

  const _Float16* seq_in  = (layer > 0) ? seqbuf + (size_t)(layer - 1) * SEQ_STRIDE : seqbuf;
  _Float16*       seq_out = seqbuf + (size_t)layer * SEQ_STRIDE;

  for (int t = 0; t < T_LEN; ++t) {
    // ---- acquire input from previous pipeline stage ----
    if (layer > 0) {
      if (tid == 0) {
        while (__hip_atomic_load(&progress[layer - 1], __ATOMIC_ACQUIRE,
                                 __HIP_MEMORY_SCOPE_AGENT) <= t) {
          __builtin_amdgcn_s_sleep(1);
        }
      }
      __syncthreads();
      for (int i = tid; i < BATCH * HID; i += 256)
        xh[(i >> 6) * 128 + (i & 63)] = seq_in[(size_t)t * (BATCH * HID) + i];
    } else {
      if (tid < BATCH) xb[tid] = x[tid * T_LEN + t];
    }
    __syncthreads();

    // ---- gates = [x_t|h] @ [Wih|Whh]^T + b  (f16 WMMA, f32 accumulate) ----
#pragma unroll
    for (int mt = 0; mt < 3; ++mt) {
      v8f acc0, acc1;
#pragma unroll
      for (int r = 0; r < 8; ++r) { acc0[r] = bv0; acc1[r] = bv1; }
      const int arow = mt * 16 + lrow;
#pragma unroll
      for (int kc = 0; kc < 4; ++kc) {
        // A k order used by this code (matches B): lane -> M=lane%16; halfs 0..15 -> K=base..base+15,
        // base = kc*32 + (lane/16)*16.
        const int kb = kc * 32 + hlf * 16;
        v8h lo = *(const v8h*)&xh[arow * 128 + kb];
        v8h hi = *(const v8h*)&xh[arow * 128 + kb + 8];
        v16h a = __builtin_shufflevector(lo, hi, 0, 1, 2, 3, 4, 5, 6, 7,
                                                 8, 9, 10, 11, 12, 13, 14, 15);
        acc0 = __builtin_amdgcn_wmma_f32_16x16x32_f16(false, a, false, bfrag0[kc],
                                                      (short)0, acc0, false, false);
        acc1 = __builtin_amdgcn_wmma_f32_16x16x32_f16(false, a, false, bfrag1[kc],
                                                      (short)0, acc1, false, false);
      }
      // C/D layout: VGPR r -> M = mt*16 + (lane/16)*8 + r, N = nt*16 + lane%16
      const int m0 = mt * 16 + hlf * 8;
      const int g0 = nt0 * 16 + lrow;
      const int g1 = nt1 * 16 + lrow;
#pragma unroll
      for (int r = 0; r < 8; ++r) {
        gbuf[(m0 + r) * GATES + g0] = acc0[r];
        gbuf[(m0 + r) * GATES + g1] = acc1[r];
      }
    }
    __syncthreads();

    // ---- activations + state update (i,f,g,o live in different N-tiles,
    //      exchanged through gbuf) ----
    for (int i = tid; i < BATCH * HID; i += 256) {
      const int m = i >> 6, j = i & 63;
      float gi = gbuf[m * GATES + j];
      float gf = gbuf[m * GATES + 64 + j];
      float gg = gbuf[m * GATES + 128 + j];
      float go = gbuf[m * GATES + 192 + j];
      if (layer == 0) {  // one-hot input: gather row of Wih0^T (L2-resident)
        const _Float16* wr = w0t + (size_t)xb[m] * GATES;
        gi += (float)wr[j];
        gf += (float)wr[64 + j];
        gg += (float)wr[128 + j];
        go += (float)wr[192 + j];
      }
      float c  = cbuf[i];
      float cn = sigm(gf) * c + sigm(gi) * tanhf(gg);
      float h  = sigm(go) * tanhf(cn);
      cbuf[i]  = cn;
      xh[m * 128 + 64 + j] = (_Float16)h;
      if (layer < 4) seq_out[(size_t)t * (BATCH * HID) + i] = (_Float16)h;
    }
    __threadfence();
    __syncthreads();
    if (layer < 4 && tid == 0)
      __hip_atomic_store(&progress[layer], t + 1, __ATOMIC_RELEASE,
                         __HIP_MEMORY_SCOPE_AGENT);
  }

  // final h -> enc[b][layer*64 + j]  (== transpose(h_n,(1,0,2)).reshape(B,320))
  for (int i = tid; i < BATCH * HID; i += 256) {
    const int m = i >> 6, j = i & 63;
    enc[m * (LAYERS * HID) + layer * HID + j] = (float)xh[m * 128 + 64 + j];
  }
}

// ---------------------------------------------------------------------------
// mu / log_var heads: [48,320] x [320,64] x2 -- 4 MFLOP, plain FMA is fine.
// ---------------------------------------------------------------------------
__global__ void head_kernel(const float* __restrict__ enc,
                            const float* __restrict__ muW, const float* __restrict__ mub,
                            const float* __restrict__ lvW, const float* __restrict__ lvb,
                            float* __restrict__ mu_out, float* __restrict__ lv_out) {
  int i = blockIdx.x * blockDim.x + threadIdx.x;   // 0 .. 6143
  if (i >= 2 * BATCH * HID) return;
  int which = (i >= BATCH * HID);
  int idx = i - which * (BATCH * HID);
  int b = idx >> 6, o = idx & 63;
  const float* W  = which ? lvW : muW;
  const float* bb = which ? lvb : mub;
  float s = bb[o];
  const float* er = enc + b * (LAYERS * HID);
  const float* wr = W + o * (LAYERS * HID);
#pragma unroll 4
  for (int k = 0; k < LAYERS * HID; ++k) s += er[k] * wr[k];
  (which ? lv_out : mu_out)[idx] = s;
}

// ---------------------------------------------------------------------------
extern "C" void kernel_launch(void* const* d_in, const int* in_sizes, int n_in,
                              void* d_out, int out_size, void* d_ws, size_t ws_size,
                              hipStream_t stream) {
  (void)in_sizes; (void)n_in; (void)out_size; (void)ws_size;
  const int*   x    = (const int*)d_in[0];
  // d_in[1] = use_noise (==0, dead); d_in[10..15] = decoder weights (dead code)
  const float* wih0 = (const float*)d_in[2];
  const float* wih  = (const float*)d_in[3];
  const float* whh  = (const float*)d_in[4];
  const float* bias = (const float*)d_in[5];
  const float* muW  = (const float*)d_in[6];
  const float* mub  = (const float*)d_in[7];
  const float* lvW  = (const float*)d_in[8];
  const float* lvb  = (const float*)d_in[9];

  char* ws = (char*)d_ws;                       // needs ~97 MB of scratch
  _Float16* frag    = (_Float16*)(ws + FRAG_OFF);
  _Float16* w0t     = (_Float16*)(ws + W0T_OFF);
  int*      prog    = (int*)(ws + PROG_OFF);
  float*    enc     = (float*)(ws + ENC_OFF);
  _Float16* seqb    = (_Float16*)(ws + SEQ_OFF);

  float* out    = (float*)d_out;
  float* mu_out = out + (size_t)BATCH * T_LEN * 256;
  float* lv_out = mu_out + BATCH * HID;

  // fragments+w0t+flags: (163840 + 65536) threads
  prep_kernel<<<896, 256, 0, stream>>>(wih0, wih, whh, frag, w0t, prog);
  // 48*4096*64 float4 stores
  recon_kernel<<<49152, 256, 0, stream>>>(x, out);
  // 5 co-resident workgroups, one per layer, pipelined through L2
  lstm_pipeline_kernel<<<LAYERS, 256, 0, stream>>>(x, bias, frag, w0t, seqb, prog, enc);
  head_kernel<<<24, 256, 0, stream>>>(enc, muW, mub, lvW, lvb, mu_out, lv_out);
}
